// param_gen_2dlaplace_simple_46394236731881
// MI455X (gfx1250) — compile-verified, matrix operand order checked by reading, not run on hardware
//
#include <hip/hip_runtime.h>
#include <hip/hip_bf16.h>
#include <hip/hip_fp16.h>

// ---------------------------------------------------------------------------
// Problem constants (from reference)
// ---------------------------------------------------------------------------
#define N_ALPHAS   2048          // K dimension of the big GEMM
#define N_GRAD     4096          // M and N dimensions of the big GEMM
#define XDIM       752           // conv output flattened (16 ch * 47 pos)
#define LEAKY      0.01f

#define K_TILES    (N_ALPHAS / 32)   // 64  k-tiles of 32
#define MN_TILES   (N_GRAD  / 16)    // 256 fragment tiles of 16

typedef __attribute__((ext_vector_type(16))) _Float16 v16h;  // one WMMA A/B fragment per lane (32B)
typedef __attribute__((ext_vector_type(8)))  float    v8f;   // one WMMA C/D fragment per lane

// ---------------------------------------------------------------------------
// K-offset mapping inside a 16x32 f16 A-fragment (CDNA5 ISA 7.12.2):
//   lanes 0-15  hold K = {0..7, 16..23}  (VGPR0..7, 2 halves each)
//   lanes 16-31 hold K = {8..15, 24..31}
// B (32x16) uses the mirrored mapping with N = lane & 15.
// ---------------------------------------------------------------------------
__device__ __forceinline__ int frag_kofs(int lane, int j) {
    return (lane < 16) ? ((j < 8) ? j : j + 8)
                       : ((j < 8) ? j + 8 : j + 16);
}

// ---------------------------------------------------------------------------
// Kernel 1: Conv1d(1->16, k=8, stride=2, valid) + bias + LeakyReLU -> x[752]
// ---------------------------------------------------------------------------
__global__ __launch_bounds__(768)
void k_conv(const float* __restrict__ in, const float* __restrict__ w,
            const float* __restrict__ bias, float* __restrict__ x) {
    int t = threadIdx.x;
    if (t >= XDIM) return;
    int c = t / 47, p = t % 47;
    float s = bias[c];
#pragma unroll
    for (int k = 0; k < 8; ++k) s += w[c * 8 + k] * in[2 * p + k];
    x[t] = (s >= 0.f) ? s : LEAKY * s;
}

// ---------------------------------------------------------------------------
// Kernel 2: three 752->2048 GEMV heads. One 256-thread block per output row.
// which = row>>11 : 0 -> par_a -> a = softplus(-s), 1 -> par_b -> b, 2 -> Ak.
// ---------------------------------------------------------------------------
__global__ __launch_bounds__(256)
void k_heads(const float* __restrict__ x,
             const float* __restrict__ wa, const float* __restrict__ ba,
             const float* __restrict__ wb, const float* __restrict__ bb,
             const float* __restrict__ wA, const float* __restrict__ bA,
             float* __restrict__ out,
             float* __restrict__ a_ws, float* __restrict__ b_ws,
             float* __restrict__ Ak_ws) {
    __shared__ float red[256];
    int row   = blockIdx.x;          // 0 .. 3*2048-1
    int which = row >> 11;
    int i     = row & (N_ALPHAS - 1);
    const float* W = (which == 0) ? wa : (which == 1) ? wb : wA;
    const float* B = (which == 0) ? ba : (which == 1) ? bb : bA;

    float s = 0.f;
    for (int j = threadIdx.x; j < XDIM; j += 256)
        s += x[j] * W[(size_t)i * XDIM + j];
    red[threadIdx.x] = s;
    __syncthreads();
#pragma unroll
    for (int off = 128; off > 0; off >>= 1) {
        if (threadIdx.x < off) red[threadIdx.x] += red[threadIdx.x + off];
        __syncthreads();
    }
    if (threadIdx.x == 0) {
        float v   = red[0] + B[i];
        float sig = 1.f / (1.f + __expf(-v));
        // -log(sigmoid(v)) = softplus(-v), numerically stable both ways
        float nl  = (v > 0.f) ? log1pf(__expf(-v)) : (-v + log1pf(__expf(v)));
        if (which == 0)      { out[i]                  = nl;  a_ws[i]  = nl;  }
        else if (which == 1) { out[N_ALPHAS + i]       = nl;  b_ws[i]  = nl;  }
        else                 { out[2 * N_ALPHAS + i]   = sig; Ak_ws[i] = sig; }
    }
}

// ---------------------------------------------------------------------------
// Kernel 3: build A-operand, fragment-packed:
//   A[g,i] = Ak_i * exp(-a_i * t1[g]),  M=g (4096), K=i (2048)
// One thread per (mt, kt, lane) writes its 16-half lane fragment (32B).
// ---------------------------------------------------------------------------
__global__ __launch_bounds__(256)
void k_packA(const float* __restrict__ a_ws, const float* __restrict__ Ak_ws,
             const float* __restrict__ t1, _Float16* __restrict__ Apack) {
    int idx  = blockIdx.x * blockDim.x + threadIdx.x;
    int lane = idx & 31;
    int kt   = (idx >> 5) & (K_TILES - 1);
    int mt   = idx >> 11;                       // 0..255
    int g    = mt * 16 + (lane & 15);
    float tg = t1[g];
    v16h frag;
#pragma unroll
    for (int j = 0; j < 16; ++j) {
        int i  = kt * 32 + frag_kofs(lane, j);
        frag[j] = (_Float16)(Ak_ws[i] * __expf(-a_ws[i] * tg));
    }
    *(v16h*)(Apack + (((size_t)mt * K_TILES + kt) * 32 + lane) * 16) = frag;
}

// ---------------------------------------------------------------------------
// Kernel 4: build B-operand, fragment-packed:  B[i,h] = exp(-b_i * t2[h])
// ---------------------------------------------------------------------------
__global__ __launch_bounds__(256)
void k_packB(const float* __restrict__ b_ws, const float* __restrict__ t2,
             _Float16* __restrict__ Bpack) {
    int idx  = blockIdx.x * blockDim.x + threadIdx.x;
    int lane = idx & 31;
    int kt   = (idx >> 5) & (K_TILES - 1);
    int nt   = idx >> 11;                       // 0..255
    int h    = nt * 16 + (lane & 15);
    float th = t2[h];
    v16h frag;
#pragma unroll
    for (int j = 0; j < 16; ++j) {
        int i  = kt * 32 + frag_kofs(lane, j);
        frag[j] = (_Float16)__expf(-b_ws[i] * th);
    }
    *(v16h*)(Bpack + (((size_t)nt * K_TILES + kt) * 32 + lane) * 16) = frag;
}

// ---------------------------------------------------------------------------
// Kernel 5: the big WMMA GEMM.
//   out[g,h] = sum_i A[g,i]*B[i,h]   (M=N=4096, K=2048, f16 in / f32 acc)
// Block = 256 threads = 8 waves laid out 4(M) x 2(N); each wave owns a 2x2
// grid of 16x16 accumulators -> block tile 128(M) x 64(N).
// ---------------------------------------------------------------------------
__global__ __launch_bounds__(256)
void k_gemm(const _Float16* __restrict__ Apack,
            const _Float16* __restrict__ Bpack,
            float* __restrict__ out) {
    int lane = threadIdx.x & 31;
    int wave = threadIdx.x >> 5;     // 0..7
    int wm   = wave >> 1;            // 0..3
    int wn   = wave & 1;             // 0..1
    int mt0  = blockIdx.y * 8 + wm * 2;   // fragment-tile indices (of 16 rows)
    int nt0  = blockIdx.x * 4 + wn * 2;

    v8f acc[2][2] = {};

    for (int kt = 0; kt < K_TILES; ++kt) {
        v16h aF[2], bF[2];
#pragma unroll
        for (int mi = 0; mi < 2; ++mi)
            aF[mi] = *(const v16h*)(Apack +
                (((size_t)(mt0 + mi) * K_TILES + kt) * 32 + lane) * 16);
#pragma unroll
        for (int ni = 0; ni < 2; ++ni)
            bF[ni] = *(const v16h*)(Bpack +
                (((size_t)(nt0 + ni) * K_TILES + kt) * 32 + lane) * 16);

        if (kt + 1 < K_TILES) {   // hint next k-tile into cache (global_prefetch_b8)
            __builtin_prefetch(Apack + (((size_t)mt0 * K_TILES + kt + 1) * 32 + lane) * 16, 0, 3);
            __builtin_prefetch(Bpack + (((size_t)nt0 * K_TILES + kt + 1) * 32 + lane) * 16, 0, 3);
        }

#pragma unroll
        for (int mi = 0; mi < 2; ++mi)
#pragma unroll
            for (int ni = 0; ni < 2; ++ni)
                acc[mi][ni] = __builtin_amdgcn_wmma_f32_16x16x32_f16(
                    /*neg_a=*/false, aF[mi], /*neg_b=*/false, bF[ni],
                    /*c_mod=*/(short)0, acc[mi][ni],
                    /*reuse_a=*/false, /*reuse_b=*/false);
    }

    // D-fragment layout: VGPR r holds row (r + 8*(lane>=16)), col (lane&15)
    int n     = lane & 15;
    int mHalf = (lane >> 4) * 8;
#pragma unroll
    for (int mi = 0; mi < 2; ++mi)
#pragma unroll
        for (int ni = 0; ni < 2; ++ni) {
            int h = (nt0 + ni) * 16 + n;
#pragma unroll
            for (int r = 0; r < 8; ++r) {
                int g = (mt0 + mi) * 16 + mHalf + r;
                out[(size_t)g * N_GRAD + h] = acc[mi][ni][r];
            }
        }
}

// ---------------------------------------------------------------------------
// Launch
// ---------------------------------------------------------------------------
extern "C" void kernel_launch(void* const* d_in, const int* in_sizes, int n_in,
                              void* d_out, int out_size, void* d_ws, size_t ws_size,
                              hipStream_t stream) {
    const float* input_r = (const float*)d_in[0];
    const float* t1      = (const float*)d_in[1];
    const float* t2      = (const float*)d_in[2];
    const float* conv_w  = (const float*)d_in[3];
    const float* conv_b  = (const float*)d_in[4];
    const float* wa      = (const float*)d_in[5];
    const float* ba      = (const float*)d_in[6];
    const float* wb      = (const float*)d_in[7];
    const float* bb      = (const float*)d_in[8];
    const float* wA      = (const float*)d_in[9];
    const float* bA      = (const float*)d_in[10];
    float* out = (float*)d_out;

    // workspace layout
    char*  ws    = (char*)d_ws;
    float* x     = (float*)ws;                 // 752 floats
    float* a_ws  = x + 1024;                   // 2048
    float* b_ws  = a_ws + N_ALPHAS;            // 2048
    float* Ak_ws = b_ws + N_ALPHAS;            // 2048
    _Float16* Apack = (_Float16*)(ws + 64 * 1024);              // 16 MiB
    _Float16* Bpack = Apack + (size_t)N_GRAD * N_ALPHAS;        // 16 MiB

    // 1. conv + leaky -> x[752]
    k_conv<<<1, 768, 0, stream>>>(input_r, conv_w, conv_b, x);

    // 2. three GEMV heads -> a, b, Ak (to d_out and workspace)
    k_heads<<<3 * N_ALPHAS, 256, 0, stream>>>(x, wa, ba, wb, bb, wA, bA,
                                              out, a_ws, b_ws, Ak_ws);

    // 3/4. fragment-packed operand builds (256*64*32 threads each)
    int packThreads = MN_TILES * K_TILES * 32;       // 524288
    k_packA<<<packThreads / 256, 256, 0, stream>>>(a_ws, Ak_ws, t1, Apack);
    k_packB<<<packThreads / 256, 256, 0, stream>>>(b_ws, t2, Bpack);

    // 5. WMMA GEMM -> out[6144 ..]
    dim3 grid(N_GRAD / 64, N_GRAD / 128);            // (64, 32)
    k_gemm<<<grid, 256, 0, stream>>>(Apack, Bpack, out + 3 * N_ALPHAS);
}
